// DenseBayesian_42210938585444
// MI455X (gfx1250) — matrix operand order repaired and checked by reading, not compiled, not run on hardware
//
#include <hip/hip_runtime.h>

// ---------------------------------------------------------------------------
// LWTA Bayesian dense layer, deterministic path:
//   out = x @ W.T + b, then per group of 4 output units keep only argmax.
// fp32 emulated via split-bf16 (hi/lo) WMMA: acc += Ah*Bh + Ah*Bl + Al*Bh.
// Operand roles: A = W tile (M = features), B = x tile (N = batch rows), so
// each lane owns 8 consecutive features of one batch row -> LWTA is in-lane,
// stores are 2x b128 per tile.
// ---------------------------------------------------------------------------

typedef __attribute__((ext_vector_type(16))) __bf16         v16bf;
typedef __attribute__((ext_vector_type(8)))  float          v8f;
typedef __attribute__((ext_vector_type(4)))  unsigned int   v4u;
typedef __attribute__((ext_vector_type(4)))  float          v4f;
typedef __attribute__((ext_vector_type(4)))  unsigned short v4us;

union Frag {
    v16bf v;
    v4u   q[2];
};

#define D_IN   256
#define D_OUT  512
#define ROWS_B 64          // batch rows per block (block covers ALL 512 features)
#define K_LDS  264         // padded LDS row stride in bf16 elements (528 B)

__device__ __forceinline__ unsigned short f2bf(float x) {
    unsigned int u = __float_as_uint(x);
    unsigned int r = u + 0x7FFFu + ((u >> 16) & 1u);   // round-to-nearest-even
    return (unsigned short)(r >> 16);
}
__device__ __forceinline__ float bf2f(unsigned short h) {
    return __uint_as_float(((unsigned int)h) << 16);
}

// ---------------------------------------------------------------------------
// Prep: split W[512][256] f32 into bf16 hi/lo planes in workspace.
// ---------------------------------------------------------------------------
__global__ void lwta_prep_w(const float* __restrict__ W,
                            unsigned short* __restrict__ Wh,
                            unsigned short* __restrict__ Wl) {
    int i = blockIdx.x * 256 + threadIdx.x;
    if (i < D_OUT * D_IN) {
        float w  = W[i];
        unsigned short h = f2bf(w);
        Wh[i] = h;
        Wl[i] = f2bf(w - bf2f(h));
    }
}

// ---------------------------------------------------------------------------
// Main fused GEMM + hard-LWTA kernel.
// Block: 256 threads (8 waves). Block tile: 512 features x 64 batch rows.
// Wave tile: 64 features (4 M-tiles) x 64 batch rows (4 N-tiles).
// ---------------------------------------------------------------------------
__global__ __launch_bounds__(256)
void lwta_gemm(const float* __restrict__ x,
               const unsigned short* __restrict__ Wh,
               const unsigned short* __restrict__ Wl,
               const float* __restrict__ bias,
               float* __restrict__ out) {
    __shared__ unsigned short xh[ROWS_B * K_LDS];
    __shared__ unsigned short xl[ROWS_B * K_LDS];

    const int tid     = threadIdx.x;
    const int lane    = tid & 31;
    const int wid     = tid >> 5;
    const int rowBase = blockIdx.x * ROWS_B;

    // ---- stage x tile into LDS as bf16 hi/lo (row-major, padded stride) ----
    // x is streamed exactly once from HBM -> non-temporal loads keep the hot
    // 512 KB W planes resident in WGP$/L2.
    for (int i = tid; i < ROWS_B * (D_IN / 4); i += 256) {
        int row = i >> 6;              // 64 float4 per row
        int c4  = (i & 63) << 2;
        v4f v = __builtin_nontemporal_load(
                    (const v4f*)(x + (size_t)(rowBase + row) * D_IN + c4));
        v4us hh, ll;
        #pragma unroll
        for (int e = 0; e < 4; ++e) {
            unsigned short h = f2bf(v[e]);
            hh[e] = h;
            ll[e] = f2bf(v[e] - bf2f(h));
        }
        *(v4us*)&xh[row * K_LDS + c4] = hh;
        *(v4us*)&xl[row * K_LDS + c4] = ll;
    }
    __syncthreads();

    const int featBase = wid * 64;      // 8 waves x 64 features = 512
    const int halfsel  = lane >> 4;     // WMMA lane half
    const int lrow     = lane & 15;
    const int aKoff    = halfsel * 8;   // A: lanes 16-31 hold K+8 / K+24
    const int bKoff    = halfsel * 8;   // B: lanes 16-31 hold K+8 / K+24

    v8f acc[4][4];                      // [mt(features)][nt(batch)]
    #pragma unroll
    for (int mt = 0; mt < 4; ++mt)
        #pragma unroll
        for (int nt = 0; nt < 4; ++nt)
            #pragma unroll
            for (int e = 0; e < 8; ++e) acc[mt][nt][e] = 0.0f;

    // ---- K loop: 8 chunks of K=32 ------------------------------------------
    for (int kc = 0; kc < 8; ++kc) {
        const int k0 = kc * 32;

        // A fragments: W rows (features), from L2-resident bf16 planes.
        Frag Ah[4], Al[4];
        #pragma unroll
        for (int mt = 0; mt < 4; ++mt) {
            const size_t wb = (size_t)(featBase + mt * 16 + lrow) * D_IN + k0 + aKoff;
            Ah[mt].q[0] = *(const v4u*)(Wh + wb);
            Ah[mt].q[1] = *(const v4u*)(Wh + wb + 16);
            Al[mt].q[0] = *(const v4u*)(Wl + wb);
            Al[mt].q[1] = *(const v4u*)(Wl + wb + 16);
        }

        #pragma unroll
        for (int nt = 0; nt < 4; ++nt) {
            // B fragment: x batch-row columns, K-contiguous in LDS.
            const int xb = (nt * 16 + lrow) * K_LDS + k0 + bKoff;
            Frag Bh, Bl;
            Bh.q[0] = *(const v4u*)&xh[xb];
            Bh.q[1] = *(const v4u*)&xh[xb + 16];
            Bl.q[0] = *(const v4u*)&xl[xb];
            Bl.q[1] = *(const v4u*)&xl[xb + 16];

            #pragma unroll
            for (int mt = 0; mt < 4; ++mt) {
                acc[mt][nt] = __builtin_amdgcn_wmma_f32_16x16x32_bf16(
                    false, Ah[mt].v, false, Bh.v, (short)0, acc[mt][nt], false, false);
                acc[mt][nt] = __builtin_amdgcn_wmma_f32_16x16x32_bf16(
                    false, Ah[mt].v, false, Bl.v, (short)0, acc[mt][nt], false, false);
                acc[mt][nt] = __builtin_amdgcn_wmma_f32_16x16x32_bf16(
                    false, Al[mt].v, false, Bh.v, (short)0, acc[mt][nt], false, false);
            }
        }
    }

    // ---- epilogue: bias + hard LWTA (argmax in groups of 4) + b128 stores --
    // C layout: VGPR r -> feature = ftile*16 + halfsel*8 + r, batch col = lrow.
    // Each lane holds 8 consecutive features of one row: LWTA is in-register.
    #pragma unroll
    for (int mt = 0; mt < 4; ++mt) {
        const int f0 = featBase + mt * 16 + halfsel * 8;
        const v4f b0 = *(const v4f*)(bias + f0);
        const v4f b1 = *(const v4f*)(bias + f0 + 4);

        #pragma unroll
        for (int nt = 0; nt < 4; ++nt) {
            const int n = rowBase + nt * 16 + lrow;
            float v[8];
            #pragma unroll
            for (int e = 0; e < 4; ++e) {
                v[e]     = acc[mt][nt][e]     + b0[e];
                v[e + 4] = acc[mt][nt][e + 4] + b1[e];
            }

            float o[8];
            #pragma unroll
            for (int g = 0; g < 8; g += 4) {
                // first-index-wins argmax over 4 competitors (jnp.argmax ties)
                float m01 = (v[g+1] > v[g+0]) ? v[g+1] : v[g+0];
                int   i01 = (v[g+1] > v[g+0]) ? 1 : 0;
                float m23 = (v[g+3] > v[g+2]) ? v[g+3] : v[g+2];
                int   i23 = (v[g+3] > v[g+2]) ? 3 : 2;
                int   arg = (m23 > m01) ? i23 : i01;
                #pragma unroll
                for (int j = 0; j < 4; ++j)
                    o[g + j] = (j == arg) ? v[g + j] : 0.0f;
            }

            float* dst = out + (size_t)n * D_OUT + f0;
            *(v4f*)dst       = (v4f){o[0], o[1], o[2], o[3]};
            *(v4f*)(dst + 4) = (v4f){o[4], o[5], o[6], o[7]};
        }
    }
}

// ---------------------------------------------------------------------------
extern "C" void kernel_launch(void* const* d_in, const int* in_sizes, int n_in,
                              void* d_out, int out_size, void* d_ws, size_t ws_size,
                              hipStream_t stream) {
    const float* x    = (const float*)d_in[0];
    const float* W    = (const float*)d_in[1];
    const float* bias = (const float*)d_in[2];
    float* out        = (float*)d_out;

    unsigned short* Wh = (unsigned short*)d_ws;
    unsigned short* Wl = Wh + (size_t)D_OUT * D_IN;

    const int N = in_sizes[0] / D_IN;   // 262144

    lwta_prep_w<<<(D_OUT * D_IN + 255) / 256, 256, 0, stream>>>(W, Wh, Wl);

    lwta_gemm<<<N / ROWS_B, 256, 0, stream>>>(x, Wh, Wl, bias, out);
}
